// MultiExpertAutoEncoder_42752104465121
// MI455X (gfx1250) — matrix operand order repaired and checked by reading, not compiled, not run on hardware
//
#include <hip/hip_runtime.h>

#define ACT_DIM 768
#define EXPERTS 16
#define EDS 1536
#define TOPK 100
#define BATCH 4096
#define NTILES (EDS / 16)      // 96 dict n-tiles
#define KSTEPS (ACT_DIM / 32)  // 24 WMMA K-steps
#define ZWORDS (EDS / 32)      // 48 z values per lane per token
#define CHUNK_TILES 8          // n-tiles per TDM chunk (one per wave)
#define NCHUNKS (NTILES / CHUNK_TILES)  // 12

typedef __attribute__((ext_vector_type(16))) __bf16 v16bf;
typedef __attribute__((ext_vector_type(8)))  float  v8f;
typedef __attribute__((ext_vector_type(4)))  unsigned int v4u;
typedef __attribute__((ext_vector_type(8)))  int v8i;
typedef __attribute__((ext_vector_type(4)))  int v4i;

static __device__ inline __bf16 f2bf(float f) { return (__bf16)f; }

// ---------------- kernel 0: zero counters ----------------
__global__ void sae_init(int* counts, int* cursor) {
  int i = threadIdx.x;
  if (i < EXPERTS) { counts[i] = 0; cursor[i] = 0; }
}

// ---------------- kernel 1: gate argmax ----------------
__global__ __launch_bounds__(256)
void sae_gate(const float* __restrict__ x, const float* __restrict__ W_gate,
              const float* __restrict__ b_gate_lin, const float* __restrict__ b_gate,
              int* __restrict__ eid, int* __restrict__ counts) {
  const int wave = threadIdx.x >> 5, lane = threadIdx.x & 31;
  const int t = blockIdx.x * 8 + wave;
  if (t >= BATCH) return;
  float acc[EXPERTS];
#pragma unroll
  for (int e = 0; e < EXPERTS; ++e) acc[e] = 0.f;
  for (int d = lane; d < ACT_DIM; d += 32) {
    float xv = x[(size_t)t * ACT_DIM + d] - b_gate[d];
#pragma unroll
    for (int e = 0; e < EXPERTS; ++e) acc[e] = fmaf(xv, W_gate[e * ACT_DIM + d], acc[e]);
  }
#pragma unroll
  for (int e = 0; e < EXPERTS; ++e) {
#pragma unroll
    for (int off = 16; off > 0; off >>= 1) acc[e] += __shfl_xor(acc[e], off, 32);
  }
  if (lane == 0) {
    int best = 0; float bv = acc[0] + b_gate_lin[0];
#pragma unroll
    for (int e = 1; e < EXPERTS; ++e) {
      float v = acc[e] + b_gate_lin[e];
      if (v > bv) { bv = v; best = e; }    // strict > keeps lowest index on ties
    }
    eid[t] = best;
    atomicAdd(&counts[best], 1);
  }
}

// ---------------- kernel 2: exclusive prefix sum ----------------
__global__ void sae_scan(const int* __restrict__ counts, int* __restrict__ offsets) {
  if (threadIdx.x == 0) {
    int run = 0;
    offsets[0] = 0;
    for (int e = 0; e < EXPERTS; ++e) { run += counts[e]; offsets[e + 1] = run; }
  }
}

// ---------------- kernel 3: scatter tokens into expert buckets ----------------
__global__ __launch_bounds__(256)
void sae_scatter(const int* __restrict__ eid, const int* __restrict__ offsets,
                 int* __restrict__ cursor, int* __restrict__ order) {
  int t = blockIdx.x * 256 + threadIdx.x;
  if (t >= BATCH) return;
  int e = eid[t];
  int pos = offsets[e] + atomicAdd(&cursor[e], 1);
  order[pos] = t;
}

// ---------------- kernel 3b: W_enc fp32 -> bf16 (one-time, L2-resident after) ----
__global__ __launch_bounds__(256)
void sae_cvt(const float* __restrict__ W, __bf16* __restrict__ Wb, int n) {
  int i = blockIdx.x * 256 + threadIdx.x;
  const int stride = gridDim.x * 256;
  for (; i < n; i += stride) Wb[i] = f2bf(W[i]);
}

// ======================================================================
// Shared device helpers for the fused kernel bodies
// ======================================================================
__device__ __forceinline__ void sae_topk_decode(
    const float* Zs, float* sAct, int* sIdx, int* sCnt,
    const float* W_dec, const float* b_dec, const int* order,
    float* out, int e, int start, int nTok, int wave, int lane) {
  // ---- per-token top-100 via bitwise radix search (z >= 0 => bits monotone) ----
  for (int rr = 0; rr < 2; ++rr) {
    const int r = wave + 8 * rr;
    if (r >= nTok) continue;
    const float* zrow = Zs + r * EDS;
    unsigned int zb[ZWORDS];
#pragma unroll
    for (int i = 0; i < ZWORDS; ++i) zb[i] = __float_as_uint(zrow[lane + 32 * i]);

    unsigned int thr = 0u;
    for (int bit = 30; bit >= 0; --bit) {
      const unsigned int cand = thr | (1u << bit);
      int cnt = 0;
#pragma unroll
      for (int i = 0; i < ZWORDS; ++i) cnt += (zb[i] >= cand) ? 1 : 0;
#pragma unroll
      for (int off = 16; off > 0; off >>= 1) cnt += __shfl_xor(cnt, off, 32);
      if (cnt >= TOPK) thr = cand;
    }
#pragma unroll
    for (int i = 0; i < ZWORDS; ++i) {  // count(>thr) < K: all of these are in
      if (zb[i] > thr) {
        int pos = atomicAdd(&sCnt[r], 1);
        if (pos < TOPK) {
          sAct[r * 128 + pos] = __uint_as_float(zb[i]);
          sIdx[r * 128 + pos] = lane + 32 * i;
        }
      }
    }
    if (thr != 0u) {  // ties fill remaining slots; zeros never contribute
#pragma unroll
      for (int i = 0; i < ZWORDS; ++i) {
        if (zb[i] == thr) {
          int pos = atomicAdd(&sCnt[r], 1);
          if (pos < TOPK) {
            sAct[r * 128 + pos] = __uint_as_float(zb[i]);
            sIdx[r * 128 + pos] = lane + 32 * i;
          }
        }
      }
    }
  }
  __syncthreads();

  // ---- decode: x_hat = sum_k act * W_dec[e][idx] + b_dec (W_dec lives in L2) ----
  for (int rr = 0; rr < 2; ++rr) {
    const int r = wave + 8 * rr;
    if (r >= nTok) continue;
    const int t = order[start + r];
    int n = sCnt[r]; if (n > TOPK) n = TOPK;
    float acc[ACT_DIM / 32];
#pragma unroll
    for (int j = 0; j < ACT_DIM / 32; ++j) acc[j] = 0.f;
    for (int i = 0; i < n; ++i) {
      const float a = sAct[r * 128 + i];
      const int row = sIdx[r * 128 + i];
      const float* wd = W_dec + ((size_t)e * EDS + row) * ACT_DIM + lane;
#pragma unroll
      for (int j = 0; j < ACT_DIM / 32; ++j) acc[j] = fmaf(a, wd[32 * j], acc[j]);
    }
    float* op = out + (size_t)t * ACT_DIM + lane;
#pragma unroll
    for (int j = 0; j < ACT_DIM / 32; ++j) op[32 * j] = acc[j] + b_dec[lane + 32 * j];
  }
}

// ======================================================================
// kernel 4a: TDM path — W_enc(bf16) tiles DMA'd into LDS by the Tensor
// Data Mover, WMMA consumes straight from LDS (no cvt in the hot loop).
// LDS: Ws bf16[128][768] @0 (196608) | Xs bf16[16][768] @196608 (24576)
//      Zs f32[16][1536] @221184 (98304)  => 319488 B of the 320KB WGP pool
// overlay (post-GEMM): sAct f32[16][128] @0, sIdx @8192, sCnt @16384
// ======================================================================
#define TDM_SMEM 319488

__global__ __launch_bounds__(256)
void sae_main_tdm(const float* __restrict__ x, const __bf16* __restrict__ Wbf,
                  const float* __restrict__ b_enc, const float* __restrict__ W_dec,
                  const float* __restrict__ b_dec, const int* __restrict__ offsets,
                  const int* __restrict__ order, float* __restrict__ out) {
  extern __shared__ char smem[];
  __bf16* Ws = (__bf16*)smem;                 // 128 dict rows x 768 bf16
  __bf16* Xs = (__bf16*)(smem + 196608);
  float*  Zs = (float*)(smem + 221184);
  float*  sAct = (float*)smem;                // overlays Ws after GEMM
  int*    sIdx = (int*)(smem + 8192);
  int*    sCnt = (int*)(smem + 16384);

  const int e = blockIdx.y;
  const int start = offsets[e] + 16 * blockIdx.x;
  int nTok = offsets[e + 1] - start;
  if (nTok <= 0) return;
  if (nTok > 16) nTok = 16;

  const int tid = threadIdx.x;
  const int wave = tid >> 5;
  const int lane = tid & 31;
  const int lh = lane >> 4;
  const int lm = lane & 15;

  // stage X tile (x - b_dec) as bf16
  {
    int r = tid >> 4;
    int t = order[start + (r < nTok ? r : 0)];
    const float* xp = x + (size_t)t * ACT_DIM;
    for (int d = (tid & 15); d < ACT_DIM; d += 16)
      Xs[r * ACT_DIM + d] = f2bf(xp[d] - b_dec[d]);
  }
  __syncthreads();

  for (int chunk = 0; chunk < NCHUNKS; ++chunk) {
    if (wave == 0) {
      // Tensor DMA descriptor (D#): 2-D bf16 tile [768 x 128] of tensor [768 x 1536]
      unsigned long long ga =
          (unsigned long long)(uintptr_t)(Wbf + ((size_t)e * EDS + chunk * 128) * ACT_DIM);
      v4u g0;
      g0[0] = 1u;                                   // count=1, user mode
      g0[1] = 0u;                                   // lds_addr = 0 (Ws base)
      g0[2] = (unsigned)(ga & 0xFFFFFFFFu);
      g0[3] = (unsigned)((ga >> 32) & 0x01FFFFFFu) | 0x80000000u;  // type=2
      v8i g1;
      g1[0] = 0x00010000;                           // wg_mask=0, data_size=2B
      g1[1] = (int)(768u << 16);                    // tensor_dim0 = 768
      g1[2] = (int)(1536u << 16);                   // tensor_dim1 = 1536
      g1[3] = (int)(768u << 16);                    // tile_dim0 = 768
      g1[4] = 128;                                  // tile_dim1 = 128, tile_dim2 = 0
      g1[5] = 768;                                  // tensor_dim0_stride = 768
      g1[6] = 0;
      g1[7] = 0;
      v4i g2 = {0, 0, 0, 0};
      v4i g3 = {0, 0, 0, 0};
#if defined(__clang_major__) && __clang_major__ >= 23
      v8i g4 = {0, 0, 0, 0, 0, 0, 0, 0};
      __builtin_amdgcn_tensor_load_to_lds(g0, g1, g2, g3, g4, 0);
#else
      __builtin_amdgcn_tensor_load_to_lds(g0, g1, g2, g3, 0);
#endif
      __builtin_amdgcn_s_wait_tensorcnt(0);
    }
    __syncthreads();   // TDM data visible to all waves

    // one n-tile per wave, 24 WMMA K-steps, operands straight from LDS
    const int nt = chunk * CHUNK_TILES + wave;
    v8f acc = {0.f, 0.f, 0.f, 0.f, 0.f, 0.f, 0.f, 0.f};
    const __bf16* wr = Ws + (wave * 16 + lm) * ACT_DIM;
#pragma unroll 4
    for (int ks = 0; ks < KSTEPS; ++ks) {
      const int kb = ks * 32;
      v16bf a;
      const __bf16* xr = Xs + lm * ACT_DIM + kb;
#pragma unroll
      for (int i = 0; i < 8; ++i) a[i] = xr[lh * 16 + i];
#pragma unroll
      for (int i = 0; i < 8; ++i) a[8 + i] = xr[lh * 16 + 8 + i];
      v16bf b;
      const __bf16* wp = wr + kb + lh * 16;
#pragma unroll
      for (int i = 0; i < 16; ++i) b[i] = wp[i];
      acc = __builtin_amdgcn_wmma_f32_16x16x32_bf16(false, a, false, b,
                                                    (short)0, acc, false, false);
    }
    const int n = nt * 16 + lm;
    const float bias = b_enc[e * EDS + n];
#pragma unroll
    for (int v = 0; v < 8; ++v) {
      float z = acc[v] + bias;
      Zs[(v + 8 * lh) * EDS + n] = z > 0.f ? z : 0.f;
    }
    __syncthreads();   // chunk consumed before next TDM overwrites Ws
  }

  if (tid < 16) sCnt[tid] = 0;  // Ws region now free for the selection overlay
  __syncthreads();
  sae_topk_decode(Zs, sAct, sIdx, sCnt, W_dec, b_dec, order, out,
                  e, start, nTok, wave, lane);
}

// ======================================================================
// kernel 4b: streaming fallback (workspace too small for bf16 W_enc copy):
// B fragments loaded f32 from global and converted in-loop.
// ======================================================================
#define SMEM_BYTES 139328

__global__ __launch_bounds__(256)
void sae_main_stream(const float* __restrict__ x, const float* __restrict__ W_enc,
                     const float* __restrict__ b_enc, const float* __restrict__ W_dec,
                     const float* __restrict__ b_dec, const int* __restrict__ offsets,
                     const int* __restrict__ order, float* __restrict__ out) {
  extern __shared__ char smem[];
  __bf16* Xs  = (__bf16*)smem;
  float*  Zs  = (float*)(smem + 24576);
  float*  sAct = (float*)(smem + 122880);
  int*    sIdx = (int*)(smem + 131072);
  int*    sCnt = (int*)(smem + 139264);

  const int e = blockIdx.y;
  const int start = offsets[e] + 16 * blockIdx.x;
  int nTok = offsets[e + 1] - start;
  if (nTok <= 0) return;
  if (nTok > 16) nTok = 16;

  const int tid = threadIdx.x;
  const int wave = tid >> 5;
  const int lane = tid & 31;
  const int lh = lane >> 4;
  const int lm = lane & 15;

  {
    int r = tid >> 4;
    int t = order[start + (r < nTok ? r : 0)];
    const float* xp = x + (size_t)t * ACT_DIM;
    for (int d = (tid & 15); d < ACT_DIM; d += 16)
      Xs[r * ACT_DIM + d] = f2bf(xp[d] - b_dec[d]);
  }
  if (tid < 16) sCnt[tid] = 0;
  __syncthreads();

  for (int nt = wave; nt < NTILES; nt += 8) {
    const float* wrow = W_enc + ((size_t)e * EDS + nt * 16 + lm) * ACT_DIM;
    if (nt + 8 < NTILES)
      __builtin_prefetch(W_enc + ((size_t)e * EDS + (nt + 8) * 16 + lm) * ACT_DIM, 0, 1);
    v8f acc = {0.f, 0.f, 0.f, 0.f, 0.f, 0.f, 0.f, 0.f};
#pragma unroll 4
    for (int ks = 0; ks < KSTEPS; ++ks) {
      const int kb = ks * 32;
      v16bf a;
      const __bf16* xr = Xs + lm * ACT_DIM + kb;
#pragma unroll
      for (int i = 0; i < 8; ++i) a[i] = xr[lh * 16 + i];
#pragma unroll
      for (int i = 0; i < 8; ++i) a[8 + i] = xr[lh * 16 + 8 + i];
      v16bf b;
      const float* wp = wrow + kb + lh * 16;
#pragma unroll
      for (int i = 0; i < 16; ++i) b[i] = f2bf(wp[i]);
      acc = __builtin_amdgcn_wmma_f32_16x16x32_bf16(false, a, false, b,
                                                    (short)0, acc, false, false);
    }
    const int n = nt * 16 + lm;
    const float bias = b_enc[e * EDS + n];
#pragma unroll
    for (int v = 0; v < 8; ++v) {
      float z = acc[v] + bias;
      Zs[(v + 8 * lh) * EDS + n] = z > 0.f ? z : 0.f;
    }
  }
  __syncthreads();
  sae_topk_decode(Zs, sAct, sIdx, sCnt, W_dec, b_dec, order, out,
                  e, start, nTok, wave, lane);
}

// ---------------- host ----------------
extern "C" void kernel_launch(void* const* d_in, const int* in_sizes, int n_in,
                              void* d_out, int out_size, void* d_ws, size_t ws_size,
                              hipStream_t stream) {
  const float* x          = (const float*)d_in[0];
  const float* W_gate     = (const float*)d_in[1];
  const float* b_gate_lin = (const float*)d_in[2];
  const float* b_gate     = (const float*)d_in[3];
  const float* W_enc      = (const float*)d_in[4];
  const float* b_enc      = (const float*)d_in[5];
  const float* W_dec      = (const float*)d_in[6];
  const float* b_dec      = (const float*)d_in[7];
  float* out = (float*)d_out;

  int* ws      = (int*)d_ws;
  int* eid     = ws;            // [4096]
  int* counts  = ws + 4096;     // [16]
  int* offsets = ws + 4112;     // [17]
  int* cursor  = ws + 4136;     // [16]
  int* order   = ws + 4160;     // [4096]
  const size_t WBF_OFF = 33280; // 128B-aligned, past the int arrays
  const size_t WBF_BYTES = (size_t)EXPERTS * EDS * ACT_DIM * 2;

  sae_init<<<1, 64, 0, stream>>>(counts, cursor);
  sae_gate<<<BATCH / 8, 256, 0, stream>>>(x, W_gate, b_gate_lin, b_gate, eid, counts);
  sae_scan<<<1, 32, 0, stream>>>(counts, offsets);
  sae_scatter<<<BATCH / 256, 256, 0, stream>>>(eid, offsets, cursor, order);

  if (ws_size >= WBF_OFF + WBF_BYTES) {
    __bf16* Wbf = (__bf16*)((char*)d_ws + WBF_OFF);
    sae_cvt<<<4096, 256, 0, stream>>>(W_enc, Wbf, EXPERTS * EDS * ACT_DIM);
    sae_main_tdm<<<dim3(BATCH / 16, EXPERTS), 256, TDM_SMEM, stream>>>(
        x, Wbf, b_enc, W_dec, b_dec, offsets, order, out);
  } else {
    sae_main_stream<<<dim3(BATCH / 16, EXPERTS), 256, SMEM_BYTES, stream>>>(
        x, W_enc, b_enc, W_dec, b_dec, offsets, order, out);
  }
}
